// MDNDAR_63840393888034
// MI455X (gfx1250) — compile-verified, matrix operand order checked by reading, not run on hardware
//
#include <hip/hip_runtime.h>
#include <hip/hip_bf16.h>

#define BSZ 32
#define TT  2048
#define INF 256
#define HH  512
#define DD  128
#define GG  8
#define GD  (GG*DD)      // 1024
#define K1  (INF + DD)   // 384
#define NWG 16
#define TPB 512
#define NCH_I 12         // K1/32  chunks for inp-GEMM
#define NCH_H 16         // HH/32  chunks for h/o GEMMs

typedef __bf16 bf16;
typedef __attribute__((ext_vector_type(16))) __bf16 v16bf;
typedef __attribute__((ext_vector_type(8)))  float  v8f;
typedef int v4i __attribute__((__vector_size__(16)));   // matches async-LDS builtin param

union FragAB { v16bf v; uint4 q[2]; };
union FragC  { v8f v; float f[8]; };

// ---------------------------------------------------------------------------
// Fragment-major ("swizzled") LDS layout. The CDNA5 16-bit A/B WMMA fragment
// for a 16x32 tile maps (lane, dword v, half b0) -> k = 16*(v>=4) +
// 8*(lane>=16) + 2*(v&3) + b0, row/col = lane&15 (ISA 7.12.2). We store every
// LDS matrix so that one lane's 16 bf16 fragment elements are contiguous:
//   elem_index = ((tileblk*nch + chunk)*32 + lane)*16 + v*2 + b0
// so a fragment load is exactly two ds_load_b128.
// ---------------------------------------------------------------------------
__device__ inline v16bf load_frag(const bf16* __restrict__ S, int blk, int nch,
                                  int ch, int lane) {
  const uint4* p = (const uint4*)S + (size_t)(((blk * nch + ch) * 32 + lane) * 2);
  FragAB f;
  f.q[0] = p[0];
  f.q[1] = p[1];
  return f.v;
}

// Inverse map for scattered writers: (row16 in 0..15, k) -> swizzled element.
__device__ inline int swz_idx(int blk, int nch, int row16, int k) {
  const int ch = k >> 5, koff = k & 31;
  const int b0 = koff & 1, pair = koff >> 1;
  const int v = (pair & 3) + ((pair >> 3) << 2);
  const int hi = (pair >> 2) & 1;
  return (((blk * nch + ch) * 32) + row16 + (hi << 4)) * 16 + (v << 1) + b0;
}

// Async cache->LDS copy (GLOBAL_LOAD_ASYNC_TO_LDS_B128, ASYNCcnt-tracked):
// no VGPR staging on the serial critical path. Falls back to a b128
// load/store round trip on toolchains without the builtin.
__device__ inline void async_copy_b128(const bf16* gsrc, bf16* ldst) {
#if __has_builtin(__builtin_amdgcn_global_load_async_to_lds_b128)
  __builtin_amdgcn_global_load_async_to_lds_b128((v4i*)gsrc, (v4i*)ldst, 0, 0);
#else
  *(uint4*)ldst = *(const uint4*)gsrc;
#endif
}
__device__ inline void async_copy_wait() {
#if __has_builtin(__builtin_amdgcn_global_load_async_to_lds_b128)
#if __has_builtin(__builtin_amdgcn_s_wait_asynccnt)
  __builtin_amdgcn_s_wait_asynccnt(0);
#else
  asm volatile("s_wait_asynccnt 0" ::: "memory");
#endif
#endif
}

// Grid-wide sync across the 16 persistent workgroups.
// Fast path: CDNA5 cluster barrier (s_barrier_signal/wait -3); NOP when the
// grid is not launched as a cluster, backed by a monotonic device-scope
// atomic counter (zeroed per launch via hipMemsetAsync).
__device__ inline void grid_sync(unsigned* ctr, unsigned target) {
  __threadfence();
  __syncthreads();
  __builtin_amdgcn_s_cluster_barrier();
  if (threadIdx.x == 0) {
    __hip_atomic_fetch_add(ctr, 1u, __ATOMIC_ACQ_REL, __HIP_MEMORY_SCOPE_AGENT);
    while (__hip_atomic_load(ctr, __ATOMIC_ACQUIRE, __HIP_MEMORY_SCOPE_AGENT) < target)
      __builtin_amdgcn_s_sleep(2);
  }
  __syncthreads();
  __threadfence();
}

#define WMMA_BF16(a, b, c) \
  __builtin_amdgcn_wmma_f32_16x16x32_bf16(false, (a), false, (b), (short)0, (c), false, false)

// LDS footprint (bf16 elems): sW_ih 12288, sW_hh 16384, sW_lin 16384,
// sW_pi 4096 (fragment-order), sW_mu 32768, sW_sg 32768, sH 16384, sIO 16384.
// f32: sPi 256, sBias 256, sPart 4096.  Total 313344 B <= 320KB/WGP.
#define LDS_BYTES (294912 + 18432)

__global__ void __launch_bounds__(TPB, 1)
mdn_rnn_persistent(const float* __restrict__ x,
                   const float* __restrict__ W_ih, const float* __restrict__ b_ih,
                   const float* __restrict__ W_hh, const float* __restrict__ b_hh,
                   const float* __restrict__ W_lin, const float* __restrict__ b_lin,
                   const float* __restrict__ W_pi,  const float* __restrict__ b_pi,
                   const float* __restrict__ W_sg,  const float* __restrict__ b_sg,
                   const float* __restrict__ W_mu,  const float* __restrict__ b_mu,
                   float* __restrict__ out_lp, float* __restrict__ out_ls,
                   float* __restrict__ out_mu,
                   bf16* __restrict__ h_glob, bf16* __restrict__ o_glob,
                   bf16* __restrict__ mu_glob, unsigned* __restrict__ bar_ctr) {
  extern __shared__ char smem_raw[];
  bf16*  sW_ih  = (bf16*)smem_raw;              // swz: 2 nblk x 12 ch
  bf16*  sW_hh  = sW_ih  + 32 * K1;             // swz: 2 nblk x 16 ch
  bf16*  sW_lin = sW_hh  + 32 * HH;             // swz: 2 nblk x 16 ch
  bf16*  sW_pi  = sW_lin + 32 * HH;             // fragment-order 8x512
  bf16*  sW_mu  = sW_pi  + GG * HH;             // swz: 4 nblk x 16 ch
  bf16*  sW_sg  = sW_mu  + 64 * HH;             // swz: 4 nblk x 16 ch
  bf16*  sH     = sW_sg  + 64 * HH;             // swz: 2 mblk x 16 ch
  bf16*  sIO    = sH     + BSZ * HH;            // swz inp (2x12) / o (2x16)
  float* sPi    = (float*)(sIO + BSZ * HH);     // 32*8
  float* sBias  = sPi    + BSZ * GG;            // 256
  float* sPart  = sBias  + 256;                 // 4*4*256

  const int tid  = threadIdx.x;
  const int lane = tid & 31;
  const int wv   = tid >> 5;            // wave 0..15
  const int wg   = blockIdx.x;          // 0..15
  const int cA   = wg * 32;             // this WG's H-column slice
  const int cD   = wg * 64;             // this WG's G*D-column slice

  // ---- one-time weight staging: f32 global -> bf16 swizzled LDS slices ----
  {
    auto stage = [&](bf16* dst, const float* src, int nblks, int nch, int kdim, int row0) {
      const int tot = nblks * nch * 32 * 16;
      for (int i = tid; i < tot; i += TPB) {
        const int j = i & 15, ln = (i >> 4) & 31, ch = (i >> 9) % nch, blk = (i >> 9) / nch;
        const int v = j >> 1, b0 = j & 1;
        const int k = (ch << 5) + ((v & 4) << 2) + ((ln & 16) ? 8 : 0) + ((v & 3) << 1) + b0;
        const int row = row0 + blk * 16 + (ln & 15);
        dst[i] = (bf16)src[(size_t)row * kdim + k];
      }
    };
    stage(sW_ih,  W_ih,  2, NCH_I, K1, cA);
    stage(sW_hh,  W_hh,  2, NCH_H, HH, cA);
    stage(sW_lin, W_lin, 2, NCH_H, HH, cA);
    stage(sW_mu,  W_mu,  4, NCH_H, HH, cD);
    stage(sW_sg,  W_sg,  4, NCH_H, HH, cD);
    // W_pi staged in fragment order so the pi dot-product reads both operands
    // stride-1: idx = ((g*16 + ch)*2 + hi)*16 + j, k = ch*32 + ((j&8)<<1) + (hi<<3) + (j&7)
    for (int i = tid; i < GG * HH; i += TPB) {
      const int j = i & 15, hi = (i >> 4) & 1, ch = (i >> 5) & 15, g = i >> 9;
      const int k = (ch << 5) + ((j & 8) << 1) + (hi << 3) + (j & 7);
      sW_pi[i] = (bf16)W_pi[g * HH + k];
    }
  }
  if (tid < 32)       sBias[tid] = b_ih[cA + tid] + b_hh[cA + tid];
  else if (tid < 64)  sBias[tid] = b_lin[cA + tid - 32];
  else if (tid < 128) sBias[tid] = b_mu[cD + tid - 64];
  else if (tid < 192) sBias[tid] = b_sg[cD + tid - 128];
  else if (tid < 200) sBias[tid] = b_pi[tid - 192];
  for (int i = tid; i < BSZ * HH; i += TPB) sH[i] = (bf16)0.0f;   // h_{-1} = 0
  for (int i = tid; i < BSZ * GG; i += TPB) sPi[i] = 0.0f;        // fb_0 = 0
  __syncthreads();

  unsigned bar = 0;

  for (int t = 0; t < TT; ++t) {
    // ---- build inp = [x_t | fb] as bf16, written swizzled ----
    bf16* sInp = sIO;
    for (int i = tid; i < BSZ * K1; i += TPB) {
      const int b = i / K1, k = i % K1;
      float v;
      if (k < INF) {
        v = x[((size_t)b * TT + t) * INF + k];
        if (t + 1 < TT) __builtin_prefetch(&x[((size_t)b * TT + t + 1) * INF + k], 0, 1);
      } else {
        const int d = k - INF;
        float acc = 0.f;
#pragma unroll
        for (int g = 0; g < GG; ++g)
          acc += sPi[b * GG + g] * (float)mu_glob[b * GD + g * DD + d];
        v = acc;
      }
      sInp[swz_idx(b >> 4, NCH_I, b & 15, k)] = (bf16)v;
    }
    __syncthreads();

    // ---- phase A: h = tanh(inp@W_ih^T + h@W_hh^T + b), K = 384 + 512 ----
    // ks 0..1 -> inp-GEMM (6 chunks each); ks 2..3 -> h-GEMM (8 chunks each).
    {
      const int tile = wv & 3, ks = wv >> 2;
      const int mblk = tile & 1, nblk = tile >> 1;
      FragC c0, c1;
#pragma unroll
      for (int r = 0; r < 8; ++r) { c0.f[r] = 0.f; c1.f[r] = 0.f; }
      if (ks < 2) {
        const int base = ks * 6;
#pragma unroll
        for (int u = 0; u < 6; u += 2) {
          v16bf a0 = load_frag(sInp, mblk, NCH_I, base + u, lane);
          v16bf w0 = load_frag(sW_ih, nblk, NCH_I, base + u, lane);
          c0.v = WMMA_BF16(a0, w0, c0.v);
          v16bf a1 = load_frag(sInp, mblk, NCH_I, base + u + 1, lane);
          v16bf w1 = load_frag(sW_ih, nblk, NCH_I, base + u + 1, lane);
          c1.v = WMMA_BF16(a1, w1, c1.v);
        }
      } else {
        const int base = (ks - 2) * 8;
#pragma unroll
        for (int u = 0; u < 8; u += 2) {
          v16bf a0 = load_frag(sH, mblk, NCH_H, base + u, lane);
          v16bf w0 = load_frag(sW_hh, nblk, NCH_H, base + u, lane);
          c0.v = WMMA_BF16(a0, w0, c0.v);
          v16bf a1 = load_frag(sH, mblk, NCH_H, base + u + 1, lane);
          v16bf w1 = load_frag(sW_hh, nblk, NCH_H, base + u + 1, lane);
          c1.v = WMMA_BF16(a1, w1, c1.v);
        }
      }
      float* dst = sPart + (tile * 4 + ks) * 256 + lane * 8;
#pragma unroll
      for (int r = 0; r < 8; ++r) dst[r] = c0.f[r] + c1.f[r];
    }
    __syncthreads();
    // reduce K-partials + bias + tanh; write h_t slice swizzled to global
    for (int idx = tid; idx < 1024; idx += TPB) {
      const int tile = idx >> 8, pos = idx & 255;
      const int pl = pos >> 3, r = pos & 7;
      float s = 0.f;
#pragma unroll
      for (int ks = 0; ks < 4; ++ks) s += sPart[(tile * 4 + ks) * 256 + pos];
      const int m = (tile & 1) * 16 + ((pl & 16) ? 8 : 0) + r;
      const int n = (tile >> 1) * 16 + (pl & 15);
      const float hv = tanhf(s + sBias[n]);
      h_glob[swz_idx(m >> 4, NCH_H, m & 15, cA + n)] = (bf16)hv;
    }
    bar++; grid_sync(bar_ctr, bar * NWG);
    // gather full (pre-swizzled) h_t into LDS: async cache->LDS B128 moves
    for (int i = tid; i < (BSZ * HH) / 8; i += TPB)
      async_copy_b128(h_glob + i * 8, sH + i * 8);
    async_copy_wait();
    __syncthreads();

    // ---- phase B: o = h@W_lin^T + b_lin, K = 512 (4 chunks per ks) ----
    {
      const int tile = wv & 3, ks = wv >> 2;
      const int mblk = tile & 1, nblk = tile >> 1;
      FragC c0, c1;
#pragma unroll
      for (int r = 0; r < 8; ++r) { c0.f[r] = 0.f; c1.f[r] = 0.f; }
      const int base = ks * 4;
#pragma unroll
      for (int u = 0; u < 4; u += 2) {
        v16bf a0 = load_frag(sH, mblk, NCH_H, base + u, lane);
        v16bf w0 = load_frag(sW_lin, nblk, NCH_H, base + u, lane);
        c0.v = WMMA_BF16(a0, w0, c0.v);
        v16bf a1 = load_frag(sH, mblk, NCH_H, base + u + 1, lane);
        v16bf w1 = load_frag(sW_lin, nblk, NCH_H, base + u + 1, lane);
        c1.v = WMMA_BF16(a1, w1, c1.v);
      }
      float* dst = sPart + (tile * 4 + ks) * 256 + lane * 8;
#pragma unroll
      for (int r = 0; r < 8; ++r) dst[r] = c0.f[r] + c1.f[r];
    }
    __syncthreads();
    for (int idx = tid; idx < 1024; idx += TPB) {
      const int tile = idx >> 8, pos = idx & 255;
      const int pl = pos >> 3, r = pos & 7;
      float s = 0.f;
#pragma unroll
      for (int ks = 0; ks < 4; ++ks) s += sPart[(tile * 4 + ks) * 256 + pos];
      const int m = (tile & 1) * 16 + ((pl & 16) ? 8 : 0) + r;
      const int n = (tile >> 1) * 16 + (pl & 15);
      o_glob[swz_idx(m >> 4, NCH_H, m & 15, cA + n)] = (bf16)(s + sBias[32 + n]);
    }
    bar++; grid_sync(bar_ctr, bar * NWG);
    bf16* sO = sIO;   // inp is dead; reuse region for full (swizzled) o_t
    for (int i = tid; i < (BSZ * HH) / 8; i += TPB)
      async_copy_b128(o_glob + i * 8, sO + i * 8);
    async_copy_wait();
    __syncthreads();

    // ---- phase C: pi logits + log_softmax (redundant per WG) ----
    // Both operands are in matching fragment order -> stride-1 b128 LDS reads.
    if (tid < BSZ * GG) {
      const int b = tid >> 3, g = tid & 7;
      const int row16 = b & 15, mb = b >> 4;
      float acc = sBias[192 + g];
      for (int ch = 0; ch < NCH_H; ++ch) {
#pragma unroll
        for (int hi = 0; hi < 2; ++hi) {
          const bf16* po = sO + (size_t)(((mb * NCH_H + ch) * 32 + row16 + (hi << 4)) * 16);
          const bf16* pw = sW_pi + (size_t)((((g * 16 + ch) << 1) + hi) * 16);
#pragma unroll
          for (int j = 0; j < 16; ++j) acc += (float)po[j] * (float)pw[j];
        }
      }
      sPart[tid] = acc;
    }
    __syncthreads();
    if (tid < BSZ) {
      const int b = tid;
      float mx = -3.0e38f;
      for (int g = 0; g < GG; ++g) mx = fmaxf(mx, sPart[b * GG + g]);
      float sum = 0.f;
      for (int g = 0; g < GG; ++g) sum += __expf(sPart[b * GG + g] - mx);
      const float lse = mx + __logf(sum);
      for (int g = 0; g < GG; ++g) {
        const float lp = sPart[b * GG + g] - lse;
        sPi[b * GG + g] = __expf(lp);
        if (wg == 0) out_lp[((size_t)b * TT + t) * GG + g] = lp;
      }
    }
    __syncthreads();

    // ---- phase D: mu & log_sigma slices (16 waves = 16 tiles, K=512) ----
    {
      const int mat = wv >> 3;              // 0 = mu, 1 = sigma
      const int tt = wv & 7;
      const int mblk = tt & 1;
      const int nblk = tt >> 1;             // 0..3 within 64-col slice
      const bf16* W = mat ? sW_sg : sW_mu;
      FragC c0, c1;
#pragma unroll
      for (int r = 0; r < 8; ++r) { c0.f[r] = 0.f; c1.f[r] = 0.f; }
#pragma unroll
      for (int ch = 0; ch < NCH_H; ch += 2) {
        v16bf a0 = load_frag(sO, mblk, NCH_H, ch, lane);
        v16bf w0 = load_frag(W, nblk, NCH_H, ch, lane);
        c0.v = WMMA_BF16(a0, w0, c0.v);
        v16bf a1 = load_frag(sO, mblk, NCH_H, ch + 1, lane);
        v16bf w1 = load_frag(W, nblk, NCH_H, ch + 1, lane);
        c1.v = WMMA_BF16(a1, w1, c1.v);
      }
      const int n_loc = nblk * 16 + (lane & 15);
      const int gd = cD + n_loc;
      const float bias = mat ? sBias[128 + n_loc] : sBias[64 + n_loc];
      float* outp = mat ? out_ls : out_mu;
#pragma unroll
      for (int r = 0; r < 8; ++r) {
        const int m = mblk * 16 + ((lane & 16) ? 8 : 0) + r;
        const float vv = c0.f[r] + c1.f[r] + bias;
        outp[((size_t)m * TT + t) * GD + gd] = vv;
        if (!mat) mu_glob[m * GD + gd] = (bf16)vv;   // feedback for t+1
      }
    }
    bar++; grid_sync(bar_ctr, bar * NWG);   // mu/pi visible before next fb
  }
}

extern "C" void kernel_launch(void* const* d_in, const int* in_sizes, int n_in,
                              void* d_out, int out_size, void* d_ws, size_t ws_size,
                              hipStream_t stream) {
  (void)in_sizes; (void)n_in; (void)out_size; (void)ws_size;
  const float* x     = (const float*)d_in[0];
  // d_in[1] = lengths (unused by the reference outputs)
  const float* W_ih  = (const float*)d_in[2];
  const float* b_ih  = (const float*)d_in[3];
  const float* W_hh  = (const float*)d_in[4];
  const float* b_hh  = (const float*)d_in[5];
  const float* W_lin = (const float*)d_in[6];
  const float* b_lin = (const float*)d_in[7];
  const float* W_pi  = (const float*)d_in[8];
  const float* b_pi  = (const float*)d_in[9];
  const float* W_sg  = (const float*)d_in[10];
  const float* b_sg  = (const float*)d_in[11];
  const float* W_mu  = (const float*)d_in[12];
  const float* b_mu  = (const float*)d_in[13];

  float* out    = (float*)d_out;
  float* out_lp = out;                                        // (B,T,G)
  float* out_ls = out_lp + (size_t)BSZ * TT * GG;             // (B,T,G,D)
  float* out_mu = out_ls + (size_t)BSZ * TT * GD;             // (B,T,G,D)

  char* ws = (char*)d_ws;
  unsigned* bar_ctr = (unsigned*)ws;                          // 256 B slot
  bf16* h_glob  = (bf16*)(ws + 256);                          // 32x512 (swz)
  bf16* o_glob  = (bf16*)(ws + 256 + (size_t)BSZ * HH * 2);   // 32x512 (swz)
  bf16* mu_glob = (bf16*)(ws + 256 + (size_t)BSZ * HH * 4);   // 32x1024 linear
  const size_t zbytes = 256 + (size_t)BSZ * HH * 4 + (size_t)BSZ * GD * 2;
  (void)hipMemsetAsync(d_ws, 0, zbytes, stream);   // zero barrier ctr + h/mu state

  mdn_rnn_persistent<<<dim3(NWG), dim3(TPB), LDS_BYTES, stream>>>(
      x, W_ih, b_ih, W_hh, b_hh, W_lin, b_lin, W_pi, b_pi,
      W_sg, b_sg, W_mu, b_mu, out_lp, out_ls, out_mu,
      h_glob, o_glob, mu_glob, bar_ctr);
}
